// StructureExtractor_4191888081429
// MI455X (gfx1250) — matrix operand order repaired and checked by reading, not compiled, not run on hardware
//
#include <hip/hip_runtime.h>
#include <hip/hip_bf16.h>

typedef __attribute__((ext_vector_type(16))) _Float16 v16h;
typedef __attribute__((ext_vector_type(8)))  _Float16 v8h;
typedef __attribute__((ext_vector_type(8)))  float    v8f;

constexpr int N_   = 8;
constexpr int L_   = 7680;
constexpr int C_   = 128;
constexpr int A_   = 128;
constexpr int DIN  = C_ + 4 * A_;      // 640
constexpr int ROWS = N_ * L_;          // 61440 rows per stream
constexpr int TOT  = 2 * ROWS;         // both streams
constexpr int MT   = 64;               // rows per workgroup tile (4 row-tiles)
constexpr int XPAD = 8;                // LDS row pad (halves): stride 324 words == 4 mod 64 banks

// ---------------------------------------------------------------------------
// Kernel 1: convert W1 (640x640) and W2 (640x128) f32 row-major -> f16
// column-major (W1T[n][k], W2T[n][k]) so WMMA B-fragments are contiguous.
// ---------------------------------------------------------------------------
__global__ void prep_w(const float* __restrict__ W1, const float* __restrict__ W2,
                       _Float16* __restrict__ W1T, _Float16* __restrict__ W2T) {
    int i = blockIdx.x * blockDim.x + threadIdx.x;
    if (i < DIN * DIN) {
        int k = i / DIN, n = i % DIN;
        W1T[(size_t)n * DIN + k] = (_Float16)W1[i];
    }
    if (i < DIN * C_) {
        int k = i / C_, n = i % C_;
        W2T[(size_t)n * DIN + k] = (_Float16)W2[i];
    }
}

// ---------------------------------------------------------------------------
// Kernel 2: build X = [feat | struct_info] rows in f16.
// One block per (stream, n, l) row; 128 threads = 128 anchors.
// ---------------------------------------------------------------------------
__global__ __launch_bounds__(128)
void prep_x(const float* __restrict__ feat0, const float* __restrict__ feat1,
            const float* __restrict__ pts0,  const float* __restrict__ pts1,
            const int*   __restrict__ ids0,  const int*   __restrict__ ids1,
            _Float16* __restrict__ X) {
    const int row = blockIdx.x;            // 0..TOT-1
    const int t   = threadIdx.x;           // anchor index 0..127
    const int st  = row / ROWS;
    const int r   = row % ROWS;
    const int nb  = r / L_;
    const int l   = r % L_;

    const float* pts  = st ? pts1  : pts0;
    const float* feat = st ? feat1 : feat0;
    const int*   ids  = st ? ids1  : ids0;

    const float px = pts[((size_t)nb * L_ + l) * 3 + 0];
    const float py = pts[((size_t)nb * L_ + l) * 3 + 1];
    const float pz = pts[((size_t)nb * L_ + l) * 3 + 2];

    const int aid = ids[nb * A_ + t];
    const float ax = pts[((size_t)nb * L_ + aid) * 3 + 0];
    const float ay = pts[((size_t)nb * L_ + aid) * 3 + 1];
    const float az = pts[((size_t)nb * L_ + aid) * 3 + 2];

    const float dx = px - ax, dy = py - ay, dz = pz - az;
    const float ds = dx * dx + dy * dy + dz * dz;

    __shared__ float red[4][A_];
    red[0][t] = fabsf(dx);
    red[1][t] = fabsf(dy);
    red[2][t] = fabsf(dz);
    red[3][t] = fabsf(ds);
    __syncthreads();
    for (int s = A_ / 2; s > 0; s >>= 1) {
        if (t < s) {
            red[0][t] += red[0][t + s];
            red[1][t] += red[1][t + s];
            red[2][t] += red[2][t + s];
            red[3][t] += red[3][t + s];
        }
        __syncthreads();
    }
    const float i0 = 1.0f / red[0][0];
    const float i1 = 1.0f / red[1][0];
    const float i2 = 1.0f / red[2][0];
    const float i3 = 1.0f / red[3][0];

    _Float16* xr = X + (size_t)row * DIN;
    xr[t] = (_Float16)feat[((size_t)nb * L_ + l) * C_ + t];   // C_ == 128 == blockDim
    xr[C_ + 0 * A_ + t] = (_Float16)(dx * i0);
    xr[C_ + 1 * A_ + t] = (_Float16)(dy * i1);
    xr[C_ + 2 * A_ + t] = (_Float16)(dz * i2);
    xr[C_ + 3 * A_ + t] = (_Float16)(ds * i3);
}

// ---------------------------------------------------------------------------
// Kernel 3: fused  out = feat + relu(X@W1+b1)@W2 + b2  for a 64-row tile.
// 8 waves (wave32). One LDS buffer holds X, then is recycled to hold H.
// Each B fragment feeds 4 WMMAs (4 row-tiles per wave).
// ---------------------------------------------------------------------------
__global__ __launch_bounds__(256)
void fused_mlp(const _Float16* __restrict__ X,
               const _Float16* __restrict__ W1T,
               const _Float16* __restrict__ W2T,
               const float* __restrict__ b1,
               const float* __restrict__ b2,
               const float* __restrict__ feat0,
               const float* __restrict__ feat1,
               float* __restrict__ out) {
    __shared__ _Float16 S[MT][DIN + XPAD];   // X tile, then recycled as H tile

    const int tid   = threadIdx.x;
    const int wave  = tid >> 5;            // 0..7 (wave32)
    const int lane  = tid & 31;
    const int lmod  = lane & 15;           // 0..15
    const int lhalf = lane >> 4;           // 0 or 1
    const size_t row0 = (size_t)blockIdx.x * MT;

    // ---- stage X tile into LDS (vectorized 16B chunks) ----
    for (int it = tid; it < MT * (DIN / 8); it += 256) {
        const int r  = it / (DIN / 8);
        const int c8 = it % (DIN / 8);
        *(v8h*)(&S[r][c8 * 8]) = *(const v8h*)(X + (row0 + r) * DIN + c8 * 8);
    }
    __syncthreads();

    // ---- GEMM1: H[64,640] = relu(X@W1 + b1) ----
    // wave w: col-tiles c = w + 8*j (j=0..4), row-tiles rt = 0..3
    v8f acc1[4][5] = {};
    for (int kb = 0; kb < DIN / 32; ++kb) {
        v16h a[4];
#pragma unroll
        for (int rt = 0; rt < 4; ++rt) {
            const _Float16* ar = &S[rt * 16 + lmod][kb * 32 + lhalf * 16];
            const v8h alo = *(const v8h*)(ar);
            const v8h ahi = *(const v8h*)(ar + 8);
            a[rt] = __builtin_shufflevector(alo, ahi,
                0, 1, 2, 3, 4, 5, 6, 7, 8, 9, 10, 11, 12, 13, 14, 15);
        }
#pragma unroll
        for (int j = 0; j < 5; ++j) {
            const int c = wave + 8 * j;
            const v16h b = *(const v16h*)(W1T + (size_t)(c * 16 + lmod) * DIN
                                              + kb * 32 + lhalf * 16);
#pragma unroll
            for (int rt = 0; rt < 4; ++rt) {
                acc1[rt][j] = __builtin_amdgcn_wmma_f32_16x16x32_f16(
                    false, a[rt], false, b, (short)0, acc1[rt][j], false, false);
            }
        }
    }
    __syncthreads();   // everyone done reading X from S

    // bias + relu -> f16 back into S (C/D layout: VGPR i -> M = i + 8*lhalf)
#pragma unroll
    for (int j = 0; j < 5; ++j) {
        const int n = (wave + 8 * j) * 16 + lmod;
        const float bias = b1[n];
#pragma unroll
        for (int rt = 0; rt < 4; ++rt) {
            const int mbase = rt * 16 + lhalf * 8;
#pragma unroll
            for (int i = 0; i < 8; ++i) {
                float h = acc1[rt][j][i] + bias;
                h = h > 0.0f ? h : 0.0f;
                S[mbase + i][n] = (_Float16)h;
            }
        }
    }
    __syncthreads();

    // ---- GEMM2: out[64,128] = H@W2 + b2 + feat ----
    // wave w: col-tile c = w, row-tiles 0..3  (B reuse x4)
    v8f acc2[4] = {};
    for (int kb = 0; kb < DIN / 32; ++kb) {
        const v16h b = *(const v16h*)(W2T + (size_t)(wave * 16 + lmod) * DIN
                                          + kb * 32 + lhalf * 16);
#pragma unroll
        for (int rt = 0; rt < 4; ++rt) {
            const _Float16* ar = &S[rt * 16 + lmod][kb * 32 + lhalf * 16];
            const v8h alo = *(const v8h*)(ar);
            const v8h ahi = *(const v8h*)(ar + 8);
            const v16h a = __builtin_shufflevector(alo, ahi,
                0, 1, 2, 3, 4, 5, 6, 7, 8, 9, 10, 11, 12, 13, 14, 15);
            acc2[rt] = __builtin_amdgcn_wmma_f32_16x16x32_f16(
                false, a, false, b, (short)0, acc2[rt], false, false);
        }
    }

    const size_t st    = row0 / (size_t)ROWS;   // tile never straddles streams
    const size_t srow0 = row0 - st * (size_t)ROWS;
    const float* feat  = st ? feat1 : feat0;
    float* outp        = out + st * (size_t)ROWS * C_;
    const int n = wave * 16 + lmod;
    const float bias = b2[n];
#pragma unroll
    for (int rt = 0; rt < 4; ++rt) {
        const int mbase = rt * 16 + lhalf * 8;
#pragma unroll
        for (int i = 0; i < 8; ++i) {
            const size_t m = srow0 + mbase + i;
            outp[m * C_ + n] = acc2[rt][i] + bias + feat[m * C_ + n];
        }
    }
}

// ---------------------------------------------------------------------------
extern "C" void kernel_launch(void* const* d_in, const int* in_sizes, int n_in,
                              void* d_out, int out_size, void* d_ws, size_t ws_size,
                              hipStream_t stream) {
    const float* feat0 = (const float*)d_in[0];
    const float* feat1 = (const float*)d_in[1];
    const float* pts0  = (const float*)d_in[2];
    const float* pts1  = (const float*)d_in[3];
    const int*   ids0  = (const int*)d_in[4];
    const int*   ids1  = (const int*)d_in[5];
    const float* W1    = (const float*)d_in[6];
    const float* b1    = (const float*)d_in[7];
    const float* W2    = (const float*)d_in[8];
    const float* b2    = (const float*)d_in[9];
    float* out = (float*)d_out;

    // workspace layout (f16): X[TOT*DIN] | W1T[DIN*DIN] | W2T[C_*DIN]
    _Float16* X   = (_Float16*)d_ws;
    _Float16* W1T = X + (size_t)TOT * DIN;
    _Float16* W2T = W1T + (size_t)DIN * DIN;

    prep_w<<<(DIN * DIN + 255) / 256, 256, 0, stream>>>(W1, W2, W1T, W2T);
    prep_x<<<TOT, 128, 0, stream>>>(feat0, feat1, pts0, pts1, ids0, ids1, X);
    fused_mlp<<<TOT / MT, 256, 0, stream>>>(X, W1T, W2T, b1, b2, feat0, feat1, out);
}
